// QRNN_75076028334120
// MI455X (gfx1250) — matrix operand order repaired and checked by reading, not compiled, not run on hardware
//
#include <hip/hip_runtime.h>
#include <hip/hip_bf16.h>

typedef __attribute__((ext_vector_type(16))) __bf16        bf16x16;
typedef __attribute__((ext_vector_type(8)))  float         f32x8;
typedef __attribute__((ext_vector_type(8)))  unsigned int  u32x8;
typedef __attribute__((ext_vector_type(4)))  unsigned int  u32x4;
typedef __attribute__((ext_vector_type(4)))  int           i32x4;
typedef __attribute__((address_space(1)))    i32x4         gas_i32x4;
typedef __attribute__((address_space(3)))    i32x4         las_i32x4;

#define B_   8
#define T_   4096
#define C_   256
#define NG   768          // 3 * units
#define M_   (B_ * T_)    // 32768 rows
#define NCH  32           // scan chunks per batch-row
#define CL   128          // chunk length (NCH*CL == T_)

__device__ __forceinline__ unsigned short f2bf(float f) {
  unsigned u = __float_as_uint(f);
  unsigned r = 0x7FFFu + ((u >> 16) & 1u);   // round to nearest even
  return (unsigned short)((u + r) >> 16);
}
__device__ __forceinline__ float fast_sigmoid(float x) {
  return __builtin_amdgcn_rcpf(1.0f + __expf(-x));
}
__device__ __forceinline__ float fast_tanh(float x) {
  float e = __expf(2.0f * x);                 // tanh = 1 - 2/(e^{2x}+1)
  return 1.0f - 2.0f * __builtin_amdgcn_rcpf(e + 1.0f);
}

// async global->LDS copy of 16 bytes (ASYNCcnt-tracked) + wait helper
__device__ __forceinline__ void async_cp16(const char* g, char* l, unsigned lds_off) {
#if __has_builtin(__builtin_amdgcn_global_load_async_to_lds_b128)
  __builtin_amdgcn_global_load_async_to_lds_b128(
      (gas_i32x4*)g, (las_i32x4*)l, 0, 0);
#else
  asm volatile("global_load_async_to_lds_b128 %0, %1, off"
               :: "v"(lds_off), "v"(g) : "memory");
#endif
}
__device__ __forceinline__ void wait_async0() {
#if __has_builtin(__builtin_amdgcn_s_wait_asynccnt)
  __builtin_amdgcn_s_wait_asynccnt(0);
#else
  asm volatile("s_wait_asynccnt 0x0" ::: "memory");
#endif
}

// ---------------- 1) x fp32 -> bf16 (row-major [M_, C_]) ----------------
__global__ void k_cvt_x(const float* __restrict__ x,
                        unsigned short* __restrict__ xb, int n4) {
  int i = blockIdx.x * blockDim.x + threadIdx.x;
  if (i >= n4) return;
  float4 v = ((const float4*)x)[i];
  uint2 p;
  p.x = (unsigned)f2bf(v.x) | ((unsigned)f2bf(v.y) << 16);
  p.y = (unsigned)f2bf(v.z) | ((unsigned)f2bf(v.w) << 16);
  ((uint2*)xb)[i] = p;
}

// ------- 2) pack weights into per-lane WMMA B-fragment layout (bf16) -------
// Effective B[k][n], k in [0,512): == kernel[k*NG + n] ([W,C] flattening).
// Fragment (kt,nt): 32x16 bf16 tile; lane<16 -> K kt*32+0..15 of col nt*16+lane,
// lane>=16 -> K kt*32+16..31. 2 K per dword, 8 dwords per lane, 1 KB per frag.
__global__ void k_pack_b(const float* __restrict__ krn,
                         unsigned int* __restrict__ bp) {
  int tid  = blockIdx.x * blockDim.x + threadIdx.x;   // 768 frags * 32 lanes
  int frag = tid >> 5;
  int lane = tid & 31;
  if (frag >= 16 * 48) return;
  int kt = frag / 48;
  int nt = frag % 48;
  int n  = nt * 16 + (lane & 15);
  int kb = kt * 32 + ((lane & 16) ? 16 : 0);
  unsigned int* dst = bp + (size_t)frag * 256 + lane * 8;
#pragma unroll
  for (int r = 0; r < 8; ++r) {
    int k0 = kb + 2 * r;
    float f0 = krn[(size_t)k0 * NG + n];
    float f1 = krn[(size_t)(k0 + 1) * NG + n];
    dst[r] = (unsigned)f2bf(f0) | ((unsigned)f2bf(f1) << 16);
  }
}

// ---------------- 3) WMMA GEMM + bias + activation ----------------
// grid (M_/128, NG/128) = (256, 6), block 256 (8 waves).
// B band (128 frags = 128 KB) async-staged to LDS; fully unrolled K loop.
__global__ void __launch_bounds__(256)
k_gemm(const unsigned short* __restrict__ xb,
       const unsigned int*  __restrict__ bp,
       const float* __restrict__ bias,
       float* __restrict__ Z, float* __restrict__ F, float* __restrict__ O) {
  __shared__ __align__(16) char smemB[128 * 1024];

  const int mb   = blockIdx.x;
  const int nb   = blockIdx.y;          // 0..5 ; gate = nb>>1 (0=z,1=f,2=o)
  const int tid  = threadIdx.x;
  const int w    = tid >> 5;
  const int lane = tid & 31;
  const int hi   = (lane & 16) ? 1 : 0;
  const int mrow = mb * 128 + w * 16 + (lane & 15);

  // ---- async-stage the B band: frags (kt 0..15, j 0..7) -> LDS slot kt*8+j
  {
    const char* gbase = (const char*)bp;
#pragma unroll 4
    for (int it = 0; it < 32; ++it) {
      unsigned ldsByte = (unsigned)(it * 256 + tid) * 16u;   // 16 B per thread
      int slot   = (int)(ldsByte >> 10);                     // 0..127
      int within = (int)(ldsByte & 1023u);
      int kt = slot >> 3, j = slot & 7;
      size_t gByte = (((size_t)(kt * 48 + nb * 8 + j)) << 10) + (size_t)within;
      async_cp16(gbase + gByte, smemB + ldsByte, ldsByte);
    }
    wait_async0();
    __syncthreads();
  }

  f32x8 acc[8] = {};

  // Window-0 rows come from m-1 (zero for t==0); window-1 rows from m.
  // A 16x32 layout (matches packed B): lanes0-15 K{0-15}, lanes16-31 K{16-31}
  const bool zeroA = ((mrow & (T_ - 1)) == 0);
  const unsigned short* aprev =
      xb + (size_t)(zeroA ? 0 : (mrow - 1)) * C_ + hi * 16;
  const unsigned short* acur = xb + (size_t)mrow * C_ + hi * 16;
  const char* lbase = smemB + (size_t)lane * 32;

#pragma unroll
  for (int kt = 0; kt < 16; ++kt) {
    const unsigned short* ap = (kt < 8 ? aprev : acur) + (kt & 7) * 32;
    u32x4 a0 = *(const u32x4*)(ap);
    u32x4 a1 = *(const u32x4*)(ap + 8);
    u32x8 av = {a0.x, a0.y, a0.z, a0.w, a1.x, a1.y, a1.z, a1.w};
    if (kt < 8 && zeroA) {
      u32x8 zv = {0u, 0u, 0u, 0u, 0u, 0u, 0u, 0u};
      av = zv;
    }
    bf16x16 a = __builtin_bit_cast(bf16x16, av);

    const char* lb = lbase + ((size_t)kt << 13);            // kt * 8 KB
    u32x8 bv[8];
#pragma unroll
    for (int j = 0; j < 8; ++j)                              // batch ds loads
      bv[j] = *(const u32x8*)(lb + ((size_t)j << 10));
#pragma unroll
    for (int j = 0; j < 8; ++j) {
      bf16x16 b = __builtin_bit_cast(bf16x16, bv[j]);
      acc[j] = __builtin_amdgcn_wmma_f32_16x16x32_bf16(
          false, a, false, b, (short)0, acc[j], false, false);
    }
  }

  const int gate = nb >> 1;
  float* outp = (gate == 0) ? Z : ((gate == 1) ? F : O);
#pragma unroll
  for (int j = 0; j < 8; ++j) {
    const int gcol = (nb & 1) * 128 + j * 16 + (lane & 15);   // col within gate
    const float bv = bias[nb * 128 + j * 16 + (lane & 15)];
#pragma unroll
    for (int r = 0; r < 8; ++r) {
      float v = acc[j][r] + bv;
      v = (gate == 0) ? fast_tanh(v) : fast_sigmoid(v);
      const int row = mb * 128 + w * 16 + hi * 8 + r;         // C/D layout
      outp[(size_t)row * C_ + gcol] = v;
    }
  }
}

// ---------------- 4) per-chunk scan reduce: (A = prod f, b = local c) -------
__global__ void k_scan_part(const float* __restrict__ F,
                            const float* __restrict__ Z,
                            float* __restrict__ CA, float* __restrict__ CB) {
  int tid   = blockIdx.x * blockDim.x + threadIdx.x;  // 65536
  int u     = tid & 255;
  int rest  = tid >> 8;
  int b     = rest & 7;
  int chunk = rest >> 3;                              // 0..31
  size_t base = ((size_t)b * T_ + (size_t)chunk * CL) * C_ + u;
  float A = 1.0f, c = 0.0f;
#pragma unroll 4
  for (int i = 0; i < CL; ++i) {
    float f = F[base + (size_t)i * C_];
    float z = Z[base + (size_t)i * C_];
    c = f * c + (1.0f - f) * z;
    A *= f;
  }
  int ci = chunk * 2048 + b * 256 + u;
  CA[ci] = A;
  CB[ci] = c;
}

// ---------------- 5) sequential chunk-prefix per channel ----------------
__global__ void k_scan_prefix(const float* __restrict__ CA,
                              const float* __restrict__ CB,
                              float* __restrict__ P) {
  int ch = blockIdx.x * blockDim.x + threadIdx.x;     // 2048 channels
  float c = 0.0f;
#pragma unroll 1
  for (int k = 0; k < NCH; ++k) {
    P[k * 2048 + ch] = c;                             // c entering chunk k
    c = CA[k * 2048 + ch] * c + CB[k * 2048 + ch];
  }
}

// ---------------- 6) apply: rerun chunk with true c0, h = sigm(o)*c --------
__global__ void k_scan_apply(const float* __restrict__ F,
                             const float* __restrict__ Z,
                             const float* __restrict__ P,
                             float* __restrict__ O) {  // holds sigmoid(o), becomes h
  int tid   = blockIdx.x * blockDim.x + threadIdx.x;
  int u     = tid & 255;
  int rest  = tid >> 8;
  int b     = rest & 7;
  int chunk = rest >> 3;
  size_t base = ((size_t)b * T_ + (size_t)chunk * CL) * C_ + u;
  float c = P[chunk * 2048 + b * 256 + u];
#pragma unroll 4
  for (int i = 0; i < CL; ++i) {
    size_t idx = base + (size_t)i * C_;
    float f = F[idx];
    float z = Z[idx];
    c = f * c + (1.0f - f) * z;
    O[idx] = O[idx] * c;
  }
}

extern "C" void kernel_launch(void* const* d_in, const int* in_sizes, int n_in,
                              void* d_out, int out_size, void* d_ws, size_t ws_size,
                              hipStream_t stream) {
  const float* x    = (const float*)d_in[0];
  const float* krn  = (const float*)d_in[1];
  const float* bias = (const float*)d_in[2];
  float* out = (float*)d_out;
  char*  ws  = (char*)d_ws;

  // workspace layout (bytes)
  unsigned short* Xbf = (unsigned short*)(ws + 0);                      // 16 MB
  unsigned int*   Bpk = (unsigned int*)(ws + (size_t)(16u << 20));      // 768 KB
  float* Z  = (float*)(ws + (size_t)(18u << 20));                       // 32 MB
  float* F  = (float*)(ws + (size_t)(50u << 20));                       // 32 MB
  float* CA = (float*)(ws + (size_t)(82u << 20));                       // 256 KB
  float* CB = (float*)(ws + (size_t)(82u << 20) + (256u << 10));        // 256 KB
  float* P  = (float*)(ws + (size_t)(82u << 20) + (512u << 10));        // 256 KB

  k_cvt_x   <<<(M_ * C_ / 4 + 255) / 256, 256, 0, stream>>>(x, Xbf, M_ * C_ / 4);
  k_pack_b  <<<96, 256, 0, stream>>>(krn, Bpk);
  k_gemm    <<<dim3(M_ / 128, NG / 128), 256, 0, stream>>>(Xbf, Bpk, bias, Z, F, out);
  k_scan_part  <<<256, 256, 0, stream>>>(F, Z, CA, CB);
  k_scan_prefix<<<8,   256, 0, stream>>>(CA, CB, P);
  k_scan_apply <<<256, 256, 0, stream>>>(F, Z, P, out);
}
